// AdvancedSelfAttention_47218870453073
// MI455X (gfx1250) — matrix operand order repaired and checked by reading, not compiled, not run on hardware
//
#include <hip/hip_runtime.h>
#include <math.h>
#include <stdint.h>

typedef __bf16 bf16_t;
typedef __attribute__((ext_vector_type(16))) __bf16 v16bf;
typedef __attribute__((ext_vector_type(8)))  __bf16 v8bf;
typedef __attribute__((ext_vector_type(8)))  float  v8f;
typedef int          v4i __attribute__((ext_vector_type(4)));
typedef unsigned int v4u __attribute__((ext_vector_type(4)));
typedef unsigned int v8u __attribute__((ext_vector_type(8)));

#define B_   2
#define S_   2048
#define H_   16
#define KVH_ 4
#define D_   64
#define DM_  1024
#define KV_  256

// ---------------------------------------------------------------------------
// CDNA5 async-to-LDS helpers (guarded; fallback = synchronous staging)
// ---------------------------------------------------------------------------
#if defined(__AMDGCN__) && defined(__has_builtin)
#  if __has_builtin(__builtin_amdgcn_global_load_async_to_lds_b128)
#    define HAVE_ASYNC_LDS 1
#  endif
#endif
#ifndef HAVE_ASYNC_LDS
#  define HAVE_ASYNC_LDS 0
#endif

static __device__ __forceinline__ void async_cp16(bf16_t* lds_dst, const bf16_t* gsrc) {
#if HAVE_ASYNC_LDS
  typedef __attribute__((address_space(1))) v4i gv4i_t;
  typedef __attribute__((address_space(3))) v4i lv4i_t;
  __builtin_amdgcn_global_load_async_to_lds_b128(
      (gv4i_t*)(uintptr_t)gsrc,
      (lv4i_t*)(uintptr_t)(unsigned)(uintptr_t)lds_dst,  // low 32 bits = LDS offset
      0, 0);
#else
  *(v8bf*)lds_dst = *(const v8bf*)gsrc;
#endif
}

static __device__ __forceinline__ void wait_async4() {
#if HAVE_ASYNC_LDS
  asm volatile("s_wait_asynccnt 0x4" ::: "memory");
#endif
}
static __device__ __forceinline__ void wait_async0() {
#if HAVE_ASYNC_LDS
  asm volatile("s_wait_asynccnt 0x0" ::: "memory");
#endif
}

static __device__ __forceinline__ void wait_tensor0() {
#if defined(__AMDGCN__)
#  if defined(__has_builtin) && __has_builtin(__builtin_amdgcn_s_wait_tensorcnt)
  __builtin_amdgcn_s_wait_tensorcnt(0);
#  else
  asm volatile("s_wait_tensorcnt 0x0" ::: "memory");
#  endif
#endif
}

static __device__ __forceinline__ v16bf cat8(v8bf lo, v8bf hi) {
  return __builtin_shufflevector(lo, hi, 0,1,2,3,4,5,6,7,8,9,10,11,12,13,14,15);
}

// ---------------------------------------------------------------------------
// Elementwise converts / transposes
// ---------------------------------------------------------------------------
__global__ void cvt_f32_bf16_kernel(const float* __restrict__ in, bf16_t* __restrict__ out, size_t n) {
  size_t i = (size_t)blockIdx.x * blockDim.x + threadIdx.x;
  if (i < n) out[i] = (__bf16)in[i];
}

// Wt[n][k] = W[k][n]  (W is [K][N] row-major f32)
__global__ void transpose_cvt_kernel(const float* __restrict__ W, bf16_t* __restrict__ Wt, int K, int N) {
  size_t i = (size_t)blockIdx.x * blockDim.x + threadIdx.x;
  if (i >= (size_t)K * N) return;
  int k = (int)(i % K);
  int n = (int)(i / K);
  Wt[i] = (__bf16)W[(size_t)k * N + n];
}

// ---------------------------------------------------------------------------
// WMMA GEMM with LDS double-buffered async staging.
// C[M][N] = A[M][K] (bf16) x Bt[N][K]^T (bf16) + bias[N], f32 out.
// Block = 128 threads (4 waves) -> 64x64 C tile; wave w -> rows w*16..w*16+15.
// Per k-step: stage 64x32 A tile + 64x32 B tile (4 KB each) into LDS.
// grid = (M/64, N/64). M,N mult of 64; K mult of 32.
// ---------------------------------------------------------------------------
__global__ void gemm_bf16_kernel(const bf16_t* __restrict__ A, const bf16_t* __restrict__ Bt,
                                 const float* __restrict__ bias, float* __restrict__ C,
                                 int M, int N, int K) {
  __shared__ bf16_t sA[2][64 * 32];
  __shared__ bf16_t sB[2][64 * 32];
  const int tid  = threadIdx.x;
  const int lane = tid & 31;
  const int wave = tid >> 5;
  const int half = lane >> 4;   // 0: lanes 0-15, 1: lanes 16-31
  const int l    = lane & 15;
  const int m0 = blockIdx.x * 64;
  const int n0 = blockIdx.y * 64;
  (void)M;

  // staging map: 64 rows x 32 cols = 256 chunks of 16B; thread t -> chunks 2t, 2t+1
  const int c0  = tid * 2;
  const int ar0 = c0 >> 2,       ao0 = (c0 & 3) * 8;
  const int ar1 = (c0 + 1) >> 2, ao1 = ((c0 + 1) & 3) * 8;

  auto stage = [&](int buf, int k) {
    async_cp16(&sA[buf][ar0 * 32 + ao0], A  + (size_t)(m0 + ar0) * K + k + ao0);
    async_cp16(&sA[buf][ar1 * 32 + ao1], A  + (size_t)(m0 + ar1) * K + k + ao1);
    async_cp16(&sB[buf][ar0 * 32 + ao0], Bt + (size_t)(n0 + ar0) * K + k + ao0);
    async_cp16(&sB[buf][ar1 * 32 + ao1], Bt + (size_t)(n0 + ar1) * K + k + ao1);
  };

  const int nk = K / 32;
  v8f acc0 = {}, acc1 = {}, acc2 = {}, acc3 = {};

  stage(0, 0);
  for (int i = 0; i < nk; ++i) {
    const int cur = i & 1;
    if (i + 1 < nk) { stage(cur ^ 1, (i + 1) * 32); wait_async4(); }
    else            { wait_async0(); }
    __syncthreads();

    const bf16_t* sa = &sA[cur][(wave * 16 + l) * 32 + half * 16];
    v16bf a = cat8(*(const v8bf*)sa, *(const v8bf*)(sa + 8));
    const bf16_t* sb = &sB[cur][l * 32 + half * 16];
    v16bf b0 = *(const v16bf*)(sb);
    v16bf b1 = *(const v16bf*)(sb + 16 * 32);
    v16bf b2 = *(const v16bf*)(sb + 32 * 32);
    v16bf b3 = *(const v16bf*)(sb + 48 * 32);
    acc0 = __builtin_amdgcn_wmma_f32_16x16x32_bf16(false, a, false, b0, (short)0, acc0, false, false);
    acc1 = __builtin_amdgcn_wmma_f32_16x16x32_bf16(false, a, false, b1, (short)0, acc1, false, false);
    acc2 = __builtin_amdgcn_wmma_f32_16x16x32_bf16(false, a, false, b2, (short)0, acc2, false, false);
    acc3 = __builtin_amdgcn_wmma_f32_16x16x32_bf16(false, a, false, b3, (short)0, acc3, false, false);
    __syncthreads();
  }

#pragma unroll
  for (int r = 0; r < 8; ++r) {
    float* crow = C + (size_t)(m0 + wave * 16 + r + half * 8) * N;
    crow[n0 +      l] = acc0[r] + bias[n0 +      l];
    crow[n0 + 16 + l] = acc1[r] + bias[n0 + 16 + l];
    crow[n0 + 32 + l] = acc2[r] + bias[n0 + 32 + l];
    crow[n0 + 48 + l] = acc3[r] + bias[n0 + 48 + l];
  }
}

// ---------------------------------------------------------------------------
// RoPE: lin [B*S][nheads*64] f32 -> out [B][nheads][S][64] bf16
// ---------------------------------------------------------------------------
__global__ void rope_kernel(const float* __restrict__ lin, bf16_t* __restrict__ out, int nheads) {
  size_t idx = (size_t)blockIdx.x * blockDim.x + threadIdx.x;
  int i = (int)(idx & 31);
  size_t r = idx >> 5;
  int h = (int)(r % nheads); r /= nheads;
  int s = (int)(r % S_);
  int b = (int)(r / S_);
  if (b >= B_) return;
  const float* row = lin + (size_t)(b * S_ + s) * (nheads * 64) + h * 64;
  float x1 = row[i];
  float x2 = row[i + 32];
  float invf = powf(10000.0f, -(float)(2 * i) / 64.0f);
  float ang = (float)s * invf;
  float c = cosf(ang), sn = sinf(ang);
  bf16_t* o = out + (((size_t)(b * nheads + h) * S_) + s) * 64;
  o[i]      = (__bf16)(x1 * c - x2 * sn);
  o[i + 32] = (__bf16)(x1 * sn + x2 * c);
}

// vlin [B*S][256] f32 -> Vt [B][KVH][64][S] bf16 (transposed for WMMA B operand)
__global__ void vconv_kernel(const float* __restrict__ vlin, bf16_t* __restrict__ Vt) {
  size_t idx = (size_t)blockIdx.x * blockDim.x + threadIdx.x;
  if (idx >= (size_t)B_ * S_ * KV_) return;
  int c = (int)(idx % KV_);
  size_t r = idx / KV_;
  int s = (int)(r % S_);
  int b = (int)(r / S_);
  int kvh = c / 64, d = c % 64;
  Vt[(((size_t)(b * KVH_ + kvh) * 64 + d) * S_) + s] = (__bf16)vlin[idx];
}

// ---------------------------------------------------------------------------
// Scores: attn_raw[b][h][q][key] = scale * (Q . K)
// grid = (S/16, S/64, B*H), block = 128.
// The 64-key x 64-d K tile (8 KB, contiguous) is pulled into LDS by the
// Tensor Data Mover once per block; all 4 waves read B fragments from LDS.
// ---------------------------------------------------------------------------
__global__ void scores_kernel(const bf16_t* __restrict__ Qb, const bf16_t* __restrict__ Kb,
                              float* __restrict__ attn) {
  __shared__ bf16_t sK[64 * 64];  // 8 KB
  const int lane = threadIdx.x & 31;
  const int wave = threadIdx.x >> 5;
  const int half = lane >> 4;
  const int l    = lane & 15;
  const int bh = blockIdx.z;
  const int b = bh / H_, h = bh % H_;
  const int kvh = h / (H_ / KVH_);
  const int q0     = blockIdx.x * 16;
  const int keyBlk = blockIdx.y * 64;

  const bf16_t* Qhead = Qb + (size_t)(b * H_   + h)   * S_ * 64;
  const bf16_t* Khead = Kb + (size_t)(b * KVH_ + kvh) * S_ * 64;
  const bf16_t* Ktile = Khead + (size_t)keyBlk * 64;  // 64 rows x 64 bf16, contiguous

#if defined(__AMDGCN__)
  if (wave == 0) {
    // Tensor DMA descriptor (D#): 1-D tile of 4096 x 2-byte elements.
    const unsigned laddr = (unsigned)(uintptr_t)&sK[0];          // LDS byte offset
    const unsigned long long ga = (unsigned long long)(uintptr_t)Ktile;
    v4u g0;
    g0[0] = 1u;                                   // count=1, is_restore=0, gather=0
    g0[1] = laddr;                                // lds_addr
    g0[2] = (unsigned)(ga & 0xffffffffu);         // global_addr[31:0]
    g0[3] = (unsigned)((ga >> 32) & 0x1ffffffu) | (2u << 30);  // global_addr[56:32] | type=2
    v8u g1;
    g1[0] = (1u << 16);      // workgroup_mask=0, data_size=1 (2B), no flags
    g1[1] = (4096u << 16);   // tensor_dim0[15:0] (bits 63:48)
    g1[2] = (1u << 16);      // tensor_dim0[31:16]=0 | tensor_dim1=1
    g1[3] = (4096u << 16);   // tensor_dim1 hi=0 | tile_dim0=4096
    g1[4] = 1u;              // tile_dim1=1, tile_dim2=0
    g1[5] = 4096u;           // tensor_dim0_stride[31:0]
    g1[6] = (4096u << 16);   // stride0 hi=0 | tensor_dim1_stride[15:0]
    g1[7] = 0u;              // tensor_dim1_stride[47:16]
    asm volatile("tensor_load_to_lds %0, %1" :: "s"(g0), "s"(g1) : "memory");
    wait_tensor0();
  }
  __syncthreads();
#else
  for (int c = (int)threadIdx.x; c < (64 * 64) / 8; c += 128)
    ((v8bf*)sK)[c] = ((const v8bf*)Ktile)[c];
  __syncthreads();
#endif

  v8f acc = {};
#pragma unroll
  for (int k = 0; k < 64; k += 32) {
    const bf16_t* ap = Qhead + (size_t)(q0 + l) * 64 + k + half * 16;
    v16bf a = cat8(*(const v8bf*)ap, *(const v8bf*)(ap + 8));
    const bf16_t* sb = &sK[(wave * 16 + l) * 64 + k + half * 16];
    v16bf bb = *(const v16bf*)sb;
    acc = __builtin_amdgcn_wmma_f32_16x16x32_bf16(false, a, false, bb, (short)0, acc, false, false);
  }
  const float scale = 0.125f;  // 1/sqrt(64)
  float* obase = attn + ((size_t)bh * S_ + q0) * S_ + keyBlk + wave * 16;
#pragma unroll
  for (int r = 0; r < 8; ++r)
    obase[(size_t)(r + half * 8) * S_ + l] = acc[r] * scale;
}

// ---------------------------------------------------------------------------
// Softmax over rows of 2048 (in place). grid = B*H*S, block = 256.
// ---------------------------------------------------------------------------
__global__ void softmax_kernel(float* __restrict__ attn) {
  __shared__ float buf[S_];
  __shared__ float red[256];
  float* p = attn + (size_t)blockIdx.x * S_;
  int t = threadIdx.x;

  float mx = -INFINITY;
  for (int i = t; i < S_; i += 256) { float v = p[i]; buf[i] = v; mx = fmaxf(mx, v); }
  red[t] = mx; __syncthreads();
  for (int s = 128; s > 0; s >>= 1) { if (t < s) red[t] = fmaxf(red[t], red[t + s]); __syncthreads(); }
  mx = red[0];
  __syncthreads();

  float sum = 0.f;
  for (int i = t; i < S_; i += 256) { float e = __expf(buf[i] - mx); buf[i] = e; sum += e; }
  red[t] = sum; __syncthreads();
  for (int s = 128; s > 0; s >>= 1) { if (t < s) red[t] += red[t + s]; __syncthreads(); }
  float inv = 1.0f / red[0];

  for (int i = t; i < S_; i += 256) p[i] = buf[i] * inv;
}

// ---------------------------------------------------------------------------
// attended = attn @ V : grid = (S/16, B*H), block = 128 (wave w -> d tile w*16)
// attn f32 [bh][S][S]; Vt bf16 [b][kvh][64][S]; out bf16 [B*S][DM]
// ---------------------------------------------------------------------------
__global__ void av_kernel(const float* __restrict__ attn, const bf16_t* __restrict__ Vt,
                          bf16_t* __restrict__ attended) {
  const int lane = threadIdx.x & 31;
  const int wave = threadIdx.x >> 5;
  const int half = lane >> 4;
  const int l    = lane & 15;
  const int bh = blockIdx.y;
  const int b = bh / H_, h = bh % H_;
  const int kvh = h / (H_ / KVH_);
  const int q0 = blockIdx.x * 16;
  const int d0 = wave * 16;

  const float*  abase = attn + ((size_t)bh * S_ + q0) * S_;
  const bf16_t* Vhead = Vt + (size_t)(b * KVH_ + kvh) * 64 * S_;

  v8f acc = {};
  for (int k = 0; k < S_; k += 32) {
    const float* ap = abase + (size_t)l * S_ + k + half * 16;
    if (k + 64 < S_) __builtin_prefetch(ap + 64, 0, 3);  // stream 536 MB of attn
    v8f lo = *(const v8f*)ap;
    v8f hi = *(const v8f*)(ap + 8);
    v16bf a;
#pragma unroll
    for (int j = 0; j < 8; ++j) { a[j] = (__bf16)lo[j]; a[j + 8] = (__bf16)hi[j]; }
    v16bf bb = *(const v16bf*)(Vhead + (size_t)(d0 + l) * S_ + k + half * 16);
    acc = __builtin_amdgcn_wmma_f32_16x16x32_bf16(false, a, false, bb, (short)0, acc, false, false);
  }
#pragma unroll
  for (int r = 0; r < 8; ++r)
    attended[((size_t)b * S_ + q0 + r + half * 8) * DM_ + h * 64 + d0 + l] = (__bf16)acc[r];
}

// ---------------------------------------------------------------------------
extern "C" void kernel_launch(void* const* d_in, const int* in_sizes, int n_in,
                              void* d_out, int out_size, void* d_ws, size_t ws_size,
                              hipStream_t stream) {
  (void)in_sizes; (void)n_in; (void)out_size; (void)ws_size;
  const float* x  = (const float*)d_in[0];
  const float* Wq = (const float*)d_in[1];
  const float* bq = (const float*)d_in[2];
  const float* Wk = (const float*)d_in[3];
  const float* bk = (const float*)d_in[4];
  const float* Wv = (const float*)d_in[5];
  const float* bv = (const float*)d_in[6];
  const float* Wo = (const float*)d_in[7];
  const float* bo = (const float*)d_in[8];

  float* out  = (float*)d_out;
  float* attn = out + (size_t)B_ * S_ * DM_;

  char* w = (char*)d_ws;
  bf16_t* xb   = (bf16_t*)w;  w += (size_t)B_ * S_ * DM_ * 2;       // 8 MB
  bf16_t* Wqt  = (bf16_t*)w;  w += (size_t)DM_ * DM_ * 2;           // 2 MB
  bf16_t* Wkt  = (bf16_t*)w;  w += (size_t)KV_ * DM_ * 2;           // 0.5 MB
  bf16_t* Wvt  = (bf16_t*)w;  w += (size_t)KV_ * DM_ * 2;           // 0.5 MB
  bf16_t* Wot  = (bf16_t*)w;  w += (size_t)DM_ * DM_ * 2;           // 2 MB
  float*  qlin = (float*)w;   w += (size_t)B_ * S_ * DM_ * 4;       // 16 MB
  float*  klin = (float*)w;   w += (size_t)B_ * S_ * KV_ * 4;       // 4 MB
  float*  vlin = (float*)w;   w += (size_t)B_ * S_ * KV_ * 4;       // 4 MB
  bf16_t* Qb   = (bf16_t*)w;  w += (size_t)B_ * H_   * S_ * 64 * 2; // 8 MB
  bf16_t* Kb   = (bf16_t*)w;  w += (size_t)B_ * KVH_ * S_ * 64 * 2; // 2 MB
  bf16_t* Vt   = (bf16_t*)w;  w += (size_t)B_ * KVH_ * 64 * S_ * 2; // 2 MB
  bf16_t* attb = (bf16_t*)w;  w += (size_t)B_ * S_ * DM_ * 2;       // 8 MB

  const int M = B_ * S_;  // 4096

  // 1) convert / transpose inputs to bf16
  cvt_f32_bf16_kernel<<<(M * DM_) / 256, 256, 0, stream>>>(x, xb, (size_t)M * DM_);
  transpose_cvt_kernel<<<(DM_ * DM_) / 256, 256, 0, stream>>>(Wq, Wqt, DM_, DM_);
  transpose_cvt_kernel<<<(DM_ * KV_) / 256, 256, 0, stream>>>(Wk, Wkt, DM_, KV_);
  transpose_cvt_kernel<<<(DM_ * KV_) / 256, 256, 0, stream>>>(Wv, Wvt, DM_, KV_);
  transpose_cvt_kernel<<<(DM_ * DM_) / 256, 256, 0, stream>>>(Wo, Wot, DM_, DM_);

  // 2) QKV projections (WMMA + async LDS pipeline)
  gemm_bf16_kernel<<<dim3(M / 64, DM_ / 64), 128, 0, stream>>>(xb, Wqt, bq, qlin, M, DM_, DM_);
  gemm_bf16_kernel<<<dim3(M / 64, KV_ / 64), 128, 0, stream>>>(xb, Wkt, bk, klin, M, KV_, DM_);
  gemm_bf16_kernel<<<dim3(M / 64, KV_ / 64), 128, 0, stream>>>(xb, Wvt, bv, vlin, M, KV_, DM_);

  // 3) RoPE + layout conversions
  rope_kernel<<<(B_ * S_ * H_   * 32) / 256, 256, 0, stream>>>(qlin, Qb, H_);
  rope_kernel<<<(B_ * S_ * KVH_ * 32) / 256, 256, 0, stream>>>(klin, Kb, KVH_);
  vconv_kernel<<<(B_ * S_ * KV_) / 256, 256, 0, stream>>>(vlin, Vt);

  // 4) scores -> attn region of d_out (TDM-staged K tiles)
  scores_kernel<<<dim3(S_ / 16, S_ / 64, B_ * H_), 128, 0, stream>>>(Qb, Kb, attn);

  // 5) softmax in place
  softmax_kernel<<<B_ * H_ * S_, 256, 0, stream>>>(attn);

  // 6) attended = attn @ V
  av_kernel<<<dim3(S_ / 16, B_ * H_), 128, 0, stream>>>(attn, Vt, attb);

  // 7) output projection -> d_out
  gemm_bf16_kernel<<<dim3(M / 64, DM_ / 64), 128, 0, stream>>>(attb, Wot, bo, out, M, DM_, DM_);
}
